// SputnikAGNN_60241211293937
// MI455X (gfx1250) — compile-verified, matrix operand order checked by reading, not run on hardware
//
#include <hip/hip_runtime.h>

// ---------------------------------------------------------------------------
// AGNN pipeline for MI455X (gfx1250, wave32):
//   h = relu(x @ W1 + b1)            -- bf16 WMMA, f32 accumulate
//   4x AGNN layer (cosine attention over fixed-degree-16 CSR rows)
//   out = h @ W2 + b2                -- bf16 WMMA, f32 accumulate
// ---------------------------------------------------------------------------

typedef __bf16 bf16_t;
typedef __attribute__((ext_vector_type(16))) bf16_t v16bf;
typedef __attribute__((ext_vector_type(8)))  float  v8f;

static __device__ __forceinline__ unsigned short f2bf(float f) {
  union { float f; unsigned u; } c; c.f = f;
  unsigned u = c.u;
  u += 0x7FFFu + ((u >> 16) & 1u);   // round-to-nearest-even
  return (unsigned short)(u >> 16);
}

union Frag {
  unsigned short u[16];
  uint4 q[2];
  v16bf v;
};

// ---------------------------------------------------------------------------
// GEMM: Y[nRows,64] = act(X[nRows,KDIM] @ W[KDIM,64] + bias)
// 256 threads = 8 waves; each wave computes a 16x64 tile with
// v_wmma_f32_16x16x32_bf16.  W is staged transposed (bf16) in LDS.
// ---------------------------------------------------------------------------
template <int KDIM, bool RELU>
__global__ __launch_bounds__(256) void gemm_bias_kernel(
    const float* __restrict__ X, const float* __restrict__ W,
    const float* __restrict__ bias, float* __restrict__ Y, int nRows) {
  __shared__ alignas(16) unsigned short Wt[64 * KDIM];  // [n][k] bf16

  const int tid = threadIdx.x;
  for (int idx = tid; idx < KDIM * 64; idx += 256) {
    int k = idx >> 6, n = idx & 63;
    Wt[n * KDIM + k] = f2bf(W[idx]);  // W is [k][n] row-major
  }
  __syncthreads();

  const int wave = tid >> 5;
  const int lane = tid & 31;
  const int lm   = lane & 15;
  const int half = lane >> 4;
  const long R = ((long)blockIdx.x * 8 + wave) * 16;
  if (R >= nRows) return;  // wave-uniform

  long r = R + lm;
  if (r >= nRows) r = nRows - 1;  // clamp; WMMA needs full EXEC

  v8f z = {0.f, 0.f, 0.f, 0.f, 0.f, 0.f, 0.f, 0.f};
  v8f acc[4];
  acc[0] = z; acc[1] = z; acc[2] = z; acc[3] = z;

  // Per-lane K groups for the 16x32 bf16 fragment layout:
  //   element e -> K = (e%8) + 8*half + 16*(e/8)
  const int g0 = 8 * half;
  const int g1 = 16 + 8 * half;

#pragma unroll
  for (int kc = 0; kc < KDIM / 32; ++kc) {
    const int kb = kc * 32;
    const float* ap = X + r * (long)KDIM + kb;
    float4 a0 = *(const float4*)(ap + g0);
    float4 a1 = *(const float4*)(ap + g0 + 4);
    float4 a2 = *(const float4*)(ap + g1);
    float4 a3 = *(const float4*)(ap + g1 + 4);
    Frag A;
    A.u[0]  = f2bf(a0.x); A.u[1]  = f2bf(a0.y); A.u[2]  = f2bf(a0.z); A.u[3]  = f2bf(a0.w);
    A.u[4]  = f2bf(a1.x); A.u[5]  = f2bf(a1.y); A.u[6]  = f2bf(a1.z); A.u[7]  = f2bf(a1.w);
    A.u[8]  = f2bf(a2.x); A.u[9]  = f2bf(a2.y); A.u[10] = f2bf(a2.z); A.u[11] = f2bf(a2.w);
    A.u[12] = f2bf(a3.x); A.u[13] = f2bf(a3.y); A.u[14] = f2bf(a3.z); A.u[15] = f2bf(a3.w);

#pragma unroll
    for (int nt = 0; nt < 4; ++nt) {
      const unsigned short* wp = &Wt[(nt * 16 + lm) * KDIM + kb];
      Frag B;
      B.q[0] = *(const uint4*)(wp + g0);  // 8 contiguous bf16 (K = kb+g0..+7)
      B.q[1] = *(const uint4*)(wp + g1);  // 8 contiguous bf16 (K = kb+g1..+7)
      acc[nt] = __builtin_amdgcn_wmma_f32_16x16x32_bf16(
          false, A.v, false, B.v, (short)0, acc[nt], false, false);
    }
  }

  float bb[4];
#pragma unroll
  for (int nt = 0; nt < 4; ++nt) bb[nt] = bias[nt * 16 + lm];

  // C/D layout: VGPR r8 -> row R + r8 + 8*half, col = nt*16 + lm
#pragma unroll
  for (int nt = 0; nt < 4; ++nt) {
#pragma unroll
    for (int r8 = 0; r8 < 8; ++r8) {
      long row = R + r8 + 8 * half;
      if (row < nRows) {
        float v = acc[nt][r8] + bb[nt];
        if (RELU) v = fmaxf(v, 0.0f);
        Y[row * 64 + nt * 16 + lm] = v;
      }
    }
  }
}

// ---------------------------------------------------------------------------
// Per-node inverse L2 norm: invn[i] = 1 / max(||h[i]||, 1e-12).
// 4 lanes per row, each reads a coalesced 64B chunk; shfl_xor pair reduce.
// ---------------------------------------------------------------------------
__global__ __launch_bounds__(256) void invnorm_kernel(
    const float* __restrict__ h, float* __restrict__ invn, int nRows) {
  int tid = blockIdx.x * 256 + threadIdx.x;
  int row = tid >> 2;
  int qtr = tid & 3;
  if (row >= nRows) return;
  const float4* p = (const float4*)(h + (long)row * 64 + qtr * 16);
  float ss = 0.f;
#pragma unroll
  for (int t = 0; t < 4; ++t) {
    float4 v = p[t];
    ss += v.x * v.x + v.y * v.y + v.z * v.z + v.w * v.w;
  }
  ss += __shfl_xor(ss, 1, 32);
  ss += __shfl_xor(ss, 2, 32);
  if (qtr == 0) invn[row] = 1.0f / fmaxf(sqrtf(ss), 1e-12f);
}

// ---------------------------------------------------------------------------
// One AGNN layer with fused relu.  One wave32 per row (degree fixed at 16,
// CSR rows contiguous -> no atomics).  Lane l owns features [2l, 2l+1], so
// each neighbor gather is one coalesced 256B row read.  Edge scores reduced
// across the wave with shfl_xor; softmax + weighted sum are lane-local.
// ---------------------------------------------------------------------------
__global__ __launch_bounds__(256) void agnn_layer_kernel(
    const float* __restrict__ hin, const float* __restrict__ invn,
    const int* __restrict__ col, float* __restrict__ hout, int nRows) {
  const int wave = threadIdx.x >> 5;
  const int lane = threadIdx.x & 31;
  int i = __builtin_amdgcn_readfirstlane((int)(blockIdx.x * 8u) + wave);
  if (i >= nRows) return;  // wave-uniform

  float2 q = *(const float2*)(hin + (long)i * 64 + 2 * lane);
  float qn = invn[i];

  int c[16];
#pragma unroll
  for (int j = 0; j < 16; ++j) c[j] = col[i * 16 + j];  // uniform -> s_load
#pragma unroll
  for (int j = 0; j < 16; ++j)
    __builtin_prefetch(hin + (long)c[j] * 64 + 2 * lane, 0, 3);

  float2 nb[16];
#pragma unroll
  for (int j = 0; j < 16; ++j)
    nb[j] = *(const float2*)(hin + (long)c[j] * 64 + 2 * lane);

  float sc[16];
#pragma unroll
  for (int j = 0; j < 16; ++j) {
    float p = q.x * nb[j].x + q.y * nb[j].y;
    p += __shfl_xor(p, 16, 32);
    p += __shfl_xor(p, 8, 32);
    p += __shfl_xor(p, 4, 32);
    p += __shfl_xor(p, 2, 32);
    p += __shfl_xor(p, 1, 32);
    sc[j] = p * qn * invn[c[j]];  // beta == 1
  }

  float m = sc[0];
#pragma unroll
  for (int j = 1; j < 16; ++j) m = fmaxf(m, sc[j]);

  float denom = 0.f;
  float2 acc; acc.x = 0.f; acc.y = 0.f;
#pragma unroll
  for (int j = 0; j < 16; ++j) {
    float ex = __expf(sc[j] - m);
    denom += ex;
    acc.x += ex * nb[j].x;
    acc.y += ex * nb[j].y;
  }
  float inv = 1.0f / fmaxf(denom, 1e-12f);
  acc.x = fmaxf(acc.x * inv, 0.0f);  // fused relu
  acc.y = fmaxf(acc.y * inv, 0.0f);
  *(float2*)(hout + (long)i * 64 + 2 * lane) = acc;
}

// ---------------------------------------------------------------------------
// Launcher.  Inputs (setup_inputs order): x, adj, row_ptr, W1, b1, W2, b2.
// adj[0..E) = col ids.  Ping-pong between ws (hA) and d_out; final GEMM2
// reads hA and writes d_out.
// ---------------------------------------------------------------------------
extern "C" void kernel_launch(void* const* d_in, const int* in_sizes, int n_in,
                              void* d_out, int out_size, void* d_ws, size_t ws_size,
                              hipStream_t stream) {
  const float* x   = (const float*)d_in[0];
  const int*   col = (const int*)d_in[1];   // adj row 0 = col ids
  const float* W1  = (const float*)d_in[3];
  const float* b1  = (const float*)d_in[4];
  const float* W2  = (const float*)d_in[5];
  const float* b2  = (const float*)d_in[6];
  float* out = (float*)d_out;

  const int N = in_sizes[0] / 256;          // 100000

  float* hA   = (float*)d_ws;               // N*64 floats
  float* invn = hA + (size_t)N * 64;        // N floats

  dim3 blk(256);
  const int gemmBlocks = (N + 127) / 128;   // 8 waves x 16 rows per block
  const int edgeBlocks = (N + 7) / 8;       // 8 rows per block
  const int normBlocks = (N * 4 + 255) / 256;

  gemm_bias_kernel<256, true><<<gemmBlocks, blk, 0, stream>>>(x, W1, b1, hA, N);

  float* bufs[2] = {hA, out};
  for (int layer = 0; layer < 4; ++layer) {
    float* hi = bufs[layer & 1];
    float* ho = bufs[(layer & 1) ^ 1];
    invnorm_kernel<<<normBlocks, blk, 0, stream>>>(hi, invn, N);
    agnn_layer_kernel<<<edgeBlocks, blk, 0, stream>>>(hi, invn, col, ho, N);
  }
  // after 4 layers the activations are back in hA
  gemm_bias_kernel<64, false><<<gemmBlocks, blk, 0, stream>>>(hA, W2, b2, out, N);
}
